// FeatureFusionModule_249108103809
// MI455X (gfx1250) — matrix operand order repaired and checked by reading, not compiled, not run on hardware
//
#include <hip/hip_runtime.h>
#include <hip/hip_bf16.h>

// FeatureFusionModule for MI455X (gfx1250).
// All GEMMs on v_wmma_f32_16x16x32_bf16 (fp32 accum). All WMMA fragment
// operands are loaded as contiguous 16B chunks (global_load_b128):
//  - weights pre-converted fp32->bf16 once (row-major [Nout,K] == B^T layout)
//  - kv projection written directly transposed (kvT[b,col,n]) so K^T V needs
//    no gathers
//  - ctx softmax output stored transposed for the q@ctx GEMM.
// Workspace requirement: ~356 MB (offsets documented in kernel_launch).

typedef __bf16 bf16;
typedef __attribute__((ext_vector_type(16))) __bf16 v16bf;
typedef __attribute__((ext_vector_type(8)))  float  v8f;
typedef __attribute__((ext_vector_type(4)))  unsigned int u32x4;
typedef __attribute__((ext_vector_type(4)))  float  f32x4;

#define DEV __device__ __forceinline__

DEV int lane_id() { return threadIdx.x & 31; }

DEV v8f wmma_bf16(v16bf a, v16bf b, v8f c) {
  // (neg_a, A, neg_b, B, c_mod, C, reuse_a, reuse_b)
  return __builtin_amdgcn_wmma_f32_16x16x32_bf16(false, a, false, b, (short)0, c, false, false);
}

// Load a 16-element bf16 fragment as two contiguous 16B chunks.
DEV v16bf ld2x16(const bf16* p0, const bf16* p1) {
  union { v16bf v; u32x4 q[2]; } f;
  f.q[0] = *(const u32x4*)p0;
  f.q[1] = *(const u32x4*)p1;
  return f.v;
}

// ---- WMMA fragment loaders (ISA 7.12.2 layouts, wave32) -------------------

// A fragment: A row-major [*, lda] bf16; tile rows m0..m0+15, k in [k0,k0+32)
// lane l<16: row=l, elems K[k0..k0+7] then K[k0+16..k0+23]; l>=16: +8 shift.
DEV v16bf load_a(const bf16* A, int lda, int m0, int k0) {
  int l = lane_id();
  const bf16* p = A + (size_t)(m0 + (l & 15)) * lda + k0 + ((l >> 4) << 3);
  return ld2x16(p, p + 16);
}

// B fragment from a bf16 matrix stored as B^T row-major [Ncols, ldb]:
// B[k,n] = BT[n*ldb + k]. lane l: col = n0+l%16, elems K[k0..k0+7] then K[k0+16..k0+23]; l>=16: +8 shift.
DEV v16bf load_bT(const bf16* BT, int ldb, int n0, int k0) {
  int l = lane_id();
  const bf16* p = BT + (size_t)(n0 + (l & 15)) * ldb + k0 + ((l >> 4) << 3);
  return ld2x16(p, p + 16);
}

// ---- Stage 0: NCHW fp32 -> [B,N,C] bf16 -----------------------------------
__global__ void k_nchw_to_bf16(const float* __restrict__ x, bf16* __restrict__ t) {
  size_t i = (size_t)blockIdx.x * blockDim.x + threadIdx.x;   // over B*C*N = 16M
  int n = (int)(i & 4095);
  int c = (int)((i >> 12) & 255);
  int b = (int)(i >> 20);
  t[(((size_t)b << 12) + n) * 256 + c] = (bf16)x[i];
}

// ---- weight fp32 -> bf16 (same layout) ------------------------------------
__global__ void k_f32_to_bf16(const float* __restrict__ s, bf16* __restrict__ d, int n) {
  int i = blockIdx.x * blockDim.x + threadIdx.x;
  if (i < n) d[i] = (bf16)s[i];
}

// ---- CrossPath projection: relu(A @ W^T + b), split into y/u --------------
// A: [M,256] bf16. W: [512,256] bf16. Wave strip: 16 rows x 64 cols.
__global__ void k_cp(const bf16* __restrict__ A, const bf16* __restrict__ W,
                     const float* __restrict__ bias,
                     bf16* __restrict__ y, bf16* __restrict__ u) {
  int strip = blockIdx.x * (blockDim.x >> 5) + (threadIdx.x >> 5);
  int m0 = (strip >> 3) << 4;          // 8 n-quads (Nout=512)
  int n0 = (strip & 7) << 6;
  v8f acc[4] = {};
  for (int k0 = 0; k0 < 256; k0 += 32) {
    v16bf a = load_a(A, 256, m0, k0);
    if (k0 + 32 < 256)   // global_prefetch_b8 for next A k-block
      __builtin_prefetch(A + (size_t)(m0 + (lane_id() & 15)) * 256 + k0 + 32, 0, 0);
#pragma unroll
    for (int j = 0; j < 4; ++j) {
      v16bf b = load_bT(W, 256, n0 + (j << 4), k0);
      acc[j] = wmma_bf16(a, b, acc[j]);
    }
  }
  int l = lane_id(), colb = l & 15, rb = (l >> 4) << 3;
#pragma unroll
  for (int j = 0; j < 4; ++j)
#pragma unroll
    for (int i = 0; i < 8; ++i) {
      int col = n0 + (j << 4) + colb;
      size_t row = (size_t)m0 + rb + i;
      float v = fmaxf(acc[j][i] + bias[col], 0.f);
      if (col < 256) y[row * 256 + col] = (bf16)v;
      else           u[row * 256 + (col - 256)] = (bf16)v;
    }
}

// ---- kv projection, epilogue written TRANSPOSED: kvT[b, col(512), n(4096)]
__global__ void k_kvT(const bf16* __restrict__ A, const bf16* __restrict__ W,
                      bf16* __restrict__ outT) {
  int strip = blockIdx.x * (blockDim.x >> 5) + (threadIdx.x >> 5);
  int m0 = (strip >> 3) << 4;
  int n0 = (strip & 7) << 6;
  v8f acc[4] = {};
  for (int k0 = 0; k0 < 256; k0 += 32) {
    v16bf a = load_a(A, 256, m0, k0);
#pragma unroll
    for (int j = 0; j < 4; ++j) {
      v16bf b = load_bT(W, 256, n0 + (j << 4), k0);
      acc[j] = wmma_bf16(a, b, acc[j]);
    }
  }
  int l = lane_id(), colb = l & 15, rb = (l >> 4) << 3;
  int b = m0 >> 12;                    // 16-row tile never crosses a batch
  int nb = (m0 & 4095) + rb;
#pragma unroll
  for (int j = 0; j < 4; ++j) {
    int col = n0 + (j << 4) + colb;
    bf16* op = outT + ((size_t)b * 512 + col) * 4096 + nb;
#pragma unroll
    for (int i = 0; i < 8; ++i) op[i] = (bf16)acc[j][i];
  }
}

// ---- ctx = scale * K^T V per (b,h): 32x32, reduction over N=4096 ----------
// 4 waves each handle a quarter of N; partials combined in LDS.
__global__ void k_ctx(const bf16* __restrict__ kvT, float* __restrict__ ctx_raw) {
  __shared__ float part[4][1024];
  int bh = blockIdx.x;                 // 128 blocks x 128 threads
  int b = bh >> 3, h = bh & 7;
  int w = threadIdx.x >> 5;
  const bf16* Kp = kvT + ((size_t)b * 512 + h * 32) * 4096;        // k^T rows
  const bf16* Vp = kvT + ((size_t)b * 512 + 256 + h * 32) * 4096;  // v^T rows
  v8f acc[2][2] = {};
  for (int n0 = w << 10; n0 < (w + 1) << 10; n0 += 32) {
    v16bf a0 = load_a(Kp, 4096, 0, n0);
    v16bf a1 = load_a(Kp, 4096, 16, n0);
    v16bf b0 = load_bT(Vp, 4096, 0, n0);
    v16bf b1 = load_bT(Vp, 4096, 16, n0);
    acc[0][0] = wmma_bf16(a0, b0, acc[0][0]);
    acc[0][1] = wmma_bf16(a0, b1, acc[0][1]);
    acc[1][0] = wmma_bf16(a1, b0, acc[1][0]);
    acc[1][1] = wmma_bf16(a1, b1, acc[1][1]);
  }
  int l = lane_id(), colb = l & 15, rb = (l >> 4) << 3;
#pragma unroll
  for (int mi = 0; mi < 2; ++mi)
#pragma unroll
    for (int j = 0; j < 2; ++j)
#pragma unroll
      for (int i = 0; i < 8; ++i)
        part[w][(mi * 16 + rb + i) * 32 + j * 16 + colb] = acc[mi][j][i];
  __syncthreads();
  const float scale = 0.17677669529663687f;   // 32^-0.5
  float* outp = ctx_raw + (size_t)bh * 1024;
  for (int e = threadIdx.x; e < 1024; e += 128)
    outp[e] = (part[0][e] + part[1][e] + part[2][e] + part[3][e]) * scale;
}

// ---- softmax over d (axis -2) of ctx[d,e]; output TRANSPOSED ctxT[e,d] ----
__global__ void k_softmax_ctx(const float* __restrict__ raw, bf16* __restrict__ outT) {
  int bh = blockIdx.x;
  int e = threadIdx.x;                     // 32 threads = columns
  const float* p = raw + (size_t)bh * 1024 + e;
  float vals[32], m = -3.0e38f;
#pragma unroll
  for (int d = 0; d < 32; ++d) { vals[d] = p[d * 32]; m = fmaxf(m, vals[d]); }
  float s = 0.f;
#pragma unroll
  for (int d = 0; d < 32; ++d) { vals[d] = __expf(vals[d] - m); s += vals[d]; }
  float r = 1.f / s;
  bf16* o = outT + (size_t)bh * 1024 + (size_t)e * 32;   // contiguous row
#pragma unroll
  for (int d = 0; d < 32; ++d) o[d] = (bf16)(vals[d] * r);
}

// ---- a = q @ ctx (per head; K=32, single WMMA per 16x16 tile) -------------
__global__ void k_attn(const bf16* __restrict__ u, const bf16* __restrict__ ctxT,
                       bf16* __restrict__ aout) {
  int strip = blockIdx.x * (blockDim.x >> 5) + (threadIdx.x >> 5);
  int bh = strip >> 8;                 // 128 (b,h) pairs
  int m0 = (strip & 255) << 4;         // 256 m-tiles of N=4096
  int b = bh >> 3, h = bh & 7;
  const bf16* Ap = u + ((size_t)b << 12) * 256 + h * 32;     // q rows, ld=256
  const bf16* Cp = ctxT + (size_t)bh * 1024;                 // [e,d] row-major
  v16bf a  = load_a(Ap, 256, m0, 0);
  v16bf b0 = load_bT(Cp, 32, 0, 0);
  v16bf b1 = load_bT(Cp, 32, 16, 0);
  v8f acc0 = {}, acc1 = {};
  acc0 = wmma_bf16(a, b0, acc0);
  acc1 = wmma_bf16(a, b1, acc1);
  int l = lane_id(), colb = l & 15, rb = (l >> 4) << 3;
  bf16* Op = aout + ((size_t)b << 12) * 256 + h * 32;
#pragma unroll
  for (int i = 0; i < 8; ++i) {
    size_t row = (size_t)m0 + rb + i;
    Op[row * 256 + colb]      = (bf16)acc0[i];
    Op[row * 256 + 16 + colb] = (bf16)acc1[i];
  }
}

// ---- ep projection + residual(t) + LayerNorm, fused (16-row tile) ---------
__global__ void k_ep_ln(const bf16* __restrict__ y, const bf16* __restrict__ a,
                        const bf16* __restrict__ W, const float* __restrict__ bias,
                        const float* __restrict__ x, const float* __restrict__ g,
                        const float* __restrict__ beta, bf16* __restrict__ o) {
  __shared__ float sm[16][257];
  __shared__ float mu[16], ri[16];
  int w = threadIdx.x >> 5;            // 4 waves, 64 cols each
  int m0 = blockIdx.x << 4;
  int n0 = w << 6;
  v8f acc[4] = {};
  for (int k0 = 0; k0 < 512; k0 += 32) {
    v16bf af = (k0 < 256) ? load_a(y, 256, m0, k0) : load_a(a, 256, m0, k0 - 256);
#pragma unroll
    for (int j = 0; j < 4; ++j) {
      v16bf bf_ = load_bT(W, 512, n0 + (j << 4), k0);
      acc[j] = wmma_bf16(af, bf_, acc[j]);
    }
  }
  int l = lane_id(), colb = l & 15, rb = (l >> 4) << 3;
  int bidx = m0 >> 12;
  int nb = (m0 & 4095) + rb;           // 8 consecutive n's per lane
#pragma unroll
  for (int j = 0; j < 4; ++j) {
    int col = n0 + (j << 4) + colb;
    const float* xp = x + (((size_t)bidx << 8) + col) * 4096 + nb;
    f32x4 x0 = *(const f32x4*)xp;
    f32x4 x1 = *(const f32x4*)(xp + 4);
    float bc = bias[col];
#pragma unroll
    for (int i = 0; i < 8; ++i) {
      float xv = (i < 4) ? x0[i & 3] : x1[i & 3];
      sm[rb + i][col] = acc[j][i] + bc + xv;
    }
  }
  __syncthreads();
  if (threadIdx.x < 16) {
    int r = threadIdx.x;
    float s = 0.f;
    for (int c = 0; c < 256; ++c) s += sm[r][c];
    float mean = s * (1.f / 256.f);
    float s2 = 0.f;
    for (int c = 0; c < 256; ++c) { float d = sm[r][c] - mean; s2 += d * d; }
    mu[r] = mean;
    ri[r] = rsqrtf(s2 * (1.f / 256.f) + 1e-5f);
  }
  __syncthreads();
  for (int e = threadIdx.x; e < 4096; e += blockDim.x) {
    int r = e >> 8, c = e & 255;
    float v = (sm[r][c] - mu[r]) * ri[r] * g[c] + beta[c];
    o[((size_t)m0 + r) * 256 + c] = (bf16)v;
  }
}

// ---- ChannelEmbed: residual 1x1 + ce1 1x1 in one pass (shared A frags) ----
__global__ void k_chembed(const bf16* __restrict__ o1, const bf16* __restrict__ o2,
                          const bf16* __restrict__ Wr, const bf16* __restrict__ We,
                          const float* __restrict__ be,
                          float* __restrict__ res, float* __restrict__ e1) {
  int strip = blockIdx.x * (blockDim.x >> 5) + (threadIdx.x >> 5);
  int m0 = (strip >> 2) << 4;
  int n0 = (strip & 3) << 6;
  v8f ar[4] = {}, ae[4] = {};
  for (int k0 = 0; k0 < 512; k0 += 32) {
    v16bf af = (k0 < 256) ? load_a(o1, 256, m0, k0) : load_a(o2, 256, m0, k0 - 256);
#pragma unroll
    for (int j = 0; j < 4; ++j) {
      v16bf br = load_bT(Wr, 512, n0 + (j << 4), k0);
      ar[j] = wmma_bf16(af, br, ar[j]);
      v16bf bb = load_bT(We, 512, n0 + (j << 4), k0);
      ae[j] = wmma_bf16(af, bb, ae[j]);
    }
  }
  int l = lane_id(), colb = l & 15, rb = (l >> 4) << 3;
#pragma unroll
  for (int j = 0; j < 4; ++j)
#pragma unroll
    for (int i = 0; i < 8; ++i) {
      int col = n0 + (j << 4) + colb;
      size_t row = (size_t)m0 + rb + i;
      res[row * 256 + col] = ar[j][i];
      e1[row * 256 + col]  = ae[j][i] + be[col];
    }
}

// ---- depthwise 3x3 + relu, bf16 out for ce3 WMMA --------------------------
__global__ void k_dwconv(const float* __restrict__ e1, const float* __restrict__ w,
                         const float* __restrict__ bias, bf16* __restrict__ out) {
  size_t i = (size_t)blockIdx.x * blockDim.x + threadIdx.x;  // over M*256
  int c = (int)(i & 255);
  size_t m = i >> 8;
  int n = (int)(m & 4095), b = (int)(m >> 12);
  int y = n >> 6, x = n & 63;
  float acc = bias[c];
#pragma unroll
  for (int dy = 0; dy < 3; ++dy) {
    int yy = y + dy - 1;
    if (yy < 0 || yy > 63) continue;
#pragma unroll
    for (int dx = 0; dx < 3; ++dx) {
      int xx = x + dx - 1;
      if (xx < 0 || xx > 63) continue;
      acc += e1[((((size_t)b << 12) + (yy << 6) + xx) << 8) + c] * w[c * 9 + dy * 3 + dx];
    }
  }
  out[i] = (bf16)fmaxf(acc, 0.f);
}

// ---- ce3 GEMM (K=256 -> Nout=256), fp32 out -------------------------------
__global__ void k_gemm_f32(const bf16* __restrict__ A, const bf16* __restrict__ W,
                           const float* __restrict__ bias, float* __restrict__ out) {
  int strip = blockIdx.x * (blockDim.x >> 5) + (threadIdx.x >> 5);
  int m0 = (strip >> 2) << 4;
  int n0 = (strip & 3) << 6;
  v8f acc[4] = {};
  for (int k0 = 0; k0 < 256; k0 += 32) {
    v16bf a = load_a(A, 256, m0, k0);
#pragma unroll
    for (int j = 0; j < 4; ++j) {
      v16bf b = load_bT(W, 256, n0 + (j << 4), k0);
      acc[j] = wmma_bf16(a, b, acc[j]);
    }
  }
  int l = lane_id(), colb = l & 15, rb = (l >> 4) << 3;
#pragma unroll
  for (int j = 0; j < 4; ++j)
#pragma unroll
    for (int i = 0; i < 8; ++i) {
      int col = n0 + (j << 4) + colb;
      out[((size_t)m0 + rb + i) * 256 + col] = acc[j][i] + bias[col];
    }
}

// ---- BatchNorm stats: deterministic two-pass column reduction -------------
__global__ void k_colstats(const float* __restrict__ X, float* __restrict__ ps,
                           float* __restrict__ pq) {
  int c = threadIdx.x;                 // 256 blocks x 256 threads, 256 rows each
  size_t r0 = (size_t)blockIdx.x * 256;
  float s = 0.f, q = 0.f;
  for (int i = 0; i < 256; ++i) {
    float v = X[(r0 + i) * 256 + c];
    s += v; q += v * v;
  }
  ps[blockIdx.x * 256 + c] = s;
  pq[blockIdx.x * 256 + c] = q;
}

__global__ void k_colreduce(const float* __restrict__ ps, const float* __restrict__ pq,
                            float* __restrict__ mean, float* __restrict__ rinv) {
  int c = threadIdx.x;
  float s = 0.f, q = 0.f;
  for (int i = 0; i < 256; ++i) { s += ps[i * 256 + c]; q += pq[i * 256 + c]; }
  float m = s * (1.f / 65536.f);
  float v = q * (1.f / 65536.f) - m * m;
  mean[c] = m;
  rinv[c] = rsqrtf(v + 1e-5f);
}

// ---- r2 = residual + BN1(e3) ----------------------------------------------
__global__ void k_bn1(const float* __restrict__ e3, const float* __restrict__ res,
                      const float* __restrict__ mean, const float* __restrict__ rinv,
                      const float* __restrict__ g, const float* __restrict__ b,
                      float* __restrict__ r2) {
  size_t i = (size_t)blockIdx.x * blockDim.x + threadIdx.x;
  int c = (int)(i & 255);
  r2[i] = res[i] + (e3[i] - mean[c]) * rinv[c] * g[c] + b[c];
}

// ---- out (NCHW) = BN2(r2) -------------------------------------------------
__global__ void k_bn2_out(const float* __restrict__ r2, const float* __restrict__ mean,
                          const float* __restrict__ rinv, const float* __restrict__ g,
                          const float* __restrict__ b, float* __restrict__ out) {
  size_t i = (size_t)blockIdx.x * blockDim.x + threadIdx.x;  // over B*C*N
  int n = (int)(i & 4095);
  int c = (int)((i >> 12) & 255);
  int bb = (int)(i >> 20);
  float v = r2[(((size_t)bb << 12) + n) * 256 + c];
  out[i] = (v - mean[c]) * rinv[c] * g[c] + b[c];
}

extern "C" void kernel_launch(void* const* d_in, const int* in_sizes, int n_in,
                              void* d_out, int out_size, void* d_ws, size_t ws_size,
                              hipStream_t stream) {
  (void)in_sizes; (void)n_in; (void)out_size; (void)ws_size;
  const float* x1 = (const float*)d_in[0];
  const float* x2 = (const float*)d_in[1];
  const float* cp1_w = (const float*)d_in[2];
  const float* cp1_b = (const float*)d_in[3];
  const float* cp2_w = (const float*)d_in[4];
  const float* cp2_b = (const float*)d_in[5];
  const float* kv1_w = (const float*)d_in[6];
  const float* kv2_w = (const float*)d_in[7];
  const float* ep1_w = (const float*)d_in[8];
  const float* ep1_b = (const float*)d_in[9];
  const float* ep2_w = (const float*)d_in[10];
  const float* ep2_b = (const float*)d_in[11];
  const float* ln1_g = (const float*)d_in[12];
  const float* ln1_b = (const float*)d_in[13];
  const float* ln2_g = (const float*)d_in[14];
  const float* ln2_b = (const float*)d_in[15];
  const float* res_w = (const float*)d_in[16];
  const float* ce1_w = (const float*)d_in[17];
  const float* ce1_b = (const float*)d_in[18];
  const float* dw_w  = (const float*)d_in[19];
  const float* dw_b  = (const float*)d_in[20];
  const float* ce3_w = (const float*)d_in[21];
  const float* ce3_b = (const float*)d_in[22];
  const float* bn1_g = (const float*)d_in[23];
  const float* bn1_b = (const float*)d_in[24];
  const float* bn2_g = (const float*)d_in[25];
  const float* bn2_b = (const float*)d_in[26];
  float* out = (float*)d_out;

  // Workspace layout (buffers reused once producers/consumers are dead):
  char* ws = (char*)d_ws;
  const size_t MBq = 1ull << 20;
  bf16* t1    = (bf16*)(ws + 0);            // 32MB (dead after cp) -> o1
  bf16* t2    = (bf16*)(ws + 32 * MBq);     // 32MB (dead after cp) -> o2
  bf16* y1    = (bf16*)(ws + 64 * MBq);     // 32MB (dead after ep) -> resb
  bf16* u1    = (bf16*)(ws + 96 * MBq);     // 32MB (dead after attn)
  bf16* y2    = (bf16*)(ws + 128 * MBq);    // 32MB (dead after ep) -> e1
  bf16* u2    = (bf16*)(ws + 160 * MBq);    // 32MB
  bf16* kv1T  = (bf16*)(ws + 192 * MBq);    // 64MB (dead after ctx) -> a1/a2/dwo
  bf16* kv2T  = (bf16*)(ws + 256 * MBq);    // 64MB (dead after ctx) -> e3
  float* c1r  = (float*)(ws + 320 * MBq);   // 512KB (dead after softmax)
  float* c2r  = (float*)(ws + 320 * MBq + 512 * 1024);
  bf16* c1T   = (bf16*)(ws + 321 * MBq);    // 256KB (dead after attn)
  bf16* c2T   = (bf16*)(ws + 321 * MBq + 256 * 1024);
  bf16* a1    = (bf16*)(ws + 192 * MBq);    // reuse kv1T
  bf16* a2    = (bf16*)(ws + 224 * MBq);
  bf16* o1    = (bf16*)(ws + 0);            // reuse t1
  bf16* o2    = (bf16*)(ws + 32 * MBq);     // reuse t2
  float* resb = (float*)(ws + 64 * MBq);    // 64MB (reuse y1/u1)
  float* e1   = (float*)(ws + 128 * MBq);   // 64MB (reuse y2/u2)
  bf16*  dwo  = (bf16*)(ws + 192 * MBq);    // 32MB (reuse a1, dead after ep)
  float* e3   = (float*)(ws + 224 * MBq);   // 64MB (reuse a2/kv2T)
  float* r2   = (float*)(ws + 288 * MBq);   // 64MB (reuse ctx area after attn)
  float* psum = (float*)(ws + 352 * MBq);   // 256KB
  float* psq  = (float*)(ws + 352 * MBq + 256 * 1024);
  float* meanb = (float*)(ws + 352 * MBq + 512 * 1024);
  float* rinvb = (float*)(ws + 352 * MBq + 512 * 1024 + 1024);
  // bf16 weights (live for whole launch) at 353MB:
  bf16* wb = (bf16*)(ws + 353 * MBq);
  bf16* cp1wb = wb;                 // 512*256
  bf16* cp2wb = cp1wb + 131072;
  bf16* kv1wb = cp2wb + 131072;
  bf16* kv2wb = kv1wb + 131072;
  bf16* ep1wb = kv2wb + 131072;     // 256*512
  bf16* ep2wb = ep1wb + 131072;
  bf16* reswb = ep2wb + 131072;     // 256*512
  bf16* ce1wb = reswb + 131072;
  bf16* ce3wb = ce1wb + 131072;     // 256*256

  // Stage 0: weights -> bf16; NCHW -> [B,N,C] bf16
  k_f32_to_bf16<<<512, 256, 0, stream>>>(cp1_w, cp1wb, 131072);
  k_f32_to_bf16<<<512, 256, 0, stream>>>(cp2_w, cp2wb, 131072);
  k_f32_to_bf16<<<512, 256, 0, stream>>>(kv1_w, kv1wb, 131072);
  k_f32_to_bf16<<<512, 256, 0, stream>>>(kv2_w, kv2wb, 131072);
  k_f32_to_bf16<<<512, 256, 0, stream>>>(ep1_w, ep1wb, 131072);
  k_f32_to_bf16<<<512, 256, 0, stream>>>(ep2_w, ep2wb, 131072);
  k_f32_to_bf16<<<512, 256, 0, stream>>>(res_w, reswb, 131072);
  k_f32_to_bf16<<<512, 256, 0, stream>>>(ce1_w, ce1wb, 131072);
  k_f32_to_bf16<<<256, 256, 0, stream>>>(ce3_w, ce3wb, 65536);
  k_nchw_to_bf16<<<65536, 256, 0, stream>>>(x1, t1);
  k_nchw_to_bf16<<<65536, 256, 0, stream>>>(x2, t2);
  // CrossPath projections (relu + chunk into y/u)
  k_cp<<<4096, 256, 0, stream>>>(t1, cp1wb, cp1_b, y1, u1);
  k_cp<<<4096, 256, 0, stream>>>(t2, cp2wb, cp2_b, y2, u2);
  // kv projections, written transposed
  k_kvT<<<4096, 256, 0, stream>>>(u1, kv1wb, kv1T);
  k_kvT<<<4096, 256, 0, stream>>>(u2, kv2wb, kv2T);
  // contexts + softmax(dim=-2), softmax output transposed
  k_ctx<<<128, 128, 0, stream>>>(kv1T, c1r);
  k_ctx<<<128, 128, 0, stream>>>(kv2T, c2r);
  k_softmax_ctx<<<128, 32, 0, stream>>>(c1r, c1T);
  k_softmax_ctx<<<128, 32, 0, stream>>>(c2r, c2T);
  // cross attention: a1 = q1 @ ctx2, a2 = q2 @ ctx1
  k_attn<<<4096, 256, 0, stream>>>(u1, c2T, a1);
  k_attn<<<4096, 256, 0, stream>>>(u2, c1T, a2);
  // end projection + residual + LayerNorm
  k_ep_ln<<<4096, 128, 0, stream>>>(y1, a1, ep1wb, ep1_b, x1, ln1_g, ln1_b, o1);
  k_ep_ln<<<4096, 128, 0, stream>>>(y2, a2, ep2wb, ep2_b, x2, ln2_g, ln2_b, o2);
  // ChannelEmbed
  k_chembed<<<2048, 256, 0, stream>>>(o1, o2, reswb, ce1wb, ce1_b, resb, e1);
  k_dwconv<<<65536, 256, 0, stream>>>(e1, dw_w, dw_b, dwo);
  k_gemm_f32<<<2048, 256, 0, stream>>>(dwo, ce3wb, ce3_b, e3);
  // BN1 (batch stats over B*H*W per channel), residual add
  k_colstats<<<256, 256, 0, stream>>>(e3, psum, psq);
  k_colreduce<<<1, 256, 0, stream>>>(psum, psq, meanb, rinvb);
  k_bn1<<<65536, 256, 0, stream>>>(e3, resb, meanb, rinvb, bn1_g, bn1_b, r2);
  // BN2 -> NCHW output
  k_colstats<<<256, 256, 0, stream>>>(r2, psum, psq);
  k_colreduce<<<1, 256, 0, stream>>>(psum, psq, meanb, rinvb);
  k_bn2_out<<<65536, 256, 0, stream>>>(r2, meanb, rinvb, bn2_g, bn2_b, out);
}
